// LocalSlidingWindowAttentionOptimized_64201171141505
// MI455X (gfx1250) — compile-verified, matrix operand order checked by reading, not run on hardware
//
#include <hip/hip_runtime.h>
#include <hip/hip_bf16.h>

#define SEQ   2048
#define EMB   1024
#define HEADS 16
#define HD    64
#define WIN   128
#define BATCH 2

typedef __attribute__((ext_vector_type(16))) __bf16       v16bf;
typedef __attribute__((ext_vector_type(8)))  float        v8f;
typedef __attribute__((ext_vector_type(8)))  unsigned int v8u;
typedef __attribute__((ext_vector_type(4)))  unsigned int v4u;
typedef __attribute__((ext_vector_type(4)))  int          v4i;
typedef __attribute__((ext_vector_type(2)))  float        v2f;

typedef __attribute__((address_space(1))) v4i gv4i;   // global int4
typedef __attribute__((address_space(3))) v4i lv4i;   // LDS int4

union Frag { v4u q[2]; v8u u; v16bf b; };

__device__ __forceinline__ unsigned short f2bf(float f) {
  unsigned int u = __builtin_bit_cast(unsigned int, f);
  u += 0x7FFFu + ((u >> 16) & 1u);          // round-to-nearest-even
  return (unsigned short)(u >> 16);
}

#if defined(__has_builtin) && __has_builtin(__builtin_amdgcn_cvt_pk_bf16_f32)
typedef __attribute__((ext_vector_type(2))) __bf16 v2bf;
__device__ __forceinline__ unsigned int pack_bf16(float a, float b) {
  v2bf r = __builtin_amdgcn_cvt_pk_bf16_f32(a, b);
  return __builtin_bit_cast(unsigned int, r);
}
#else
__device__ __forceinline__ unsigned int pack_bf16(float a, float b) {
  return (unsigned int)f2bf(a) | ((unsigned int)f2bf(b) << 16);
}
#endif

__device__ __forceinline__ v8f wmma_bf16(const Frag& a, const Frag& b, v8f c) {
  return __builtin_amdgcn_wmma_f32_16x16x32_bf16(false, a.b, false, b.b,
                                                 (short)0, c, false, false);
}

static __device__ const v8f kZero8 = {0.f,0.f,0.f,0.f,0.f,0.f,0.f,0.f};

// ---------------------------------------------------------------------------
// Kernel 1 (fused): {Q,K,V} = X @ W{q,k,v} + b.  f32 in; Q,K -> bf16 [B,H,N,D],
// V -> bf16 transposed [B,H,D,N] (so attention P@V B-frags are dword loads).
// block = 128 threads (4 waves), tile = 64(M) x 64(N), K-step = 32, 12 WMMA/step.
// ---------------------------------------------------------------------------
__global__ __launch_bounds__(128) void qkv_gemm(const float* __restrict__ X,
                                                const float* __restrict__ Wq,
                                                const float* __restrict__ Wk,
                                                const float* __restrict__ Wv,
                                                const float* __restrict__ bq,
                                                const float* __restrict__ bk,
                                                const float* __restrict__ bv,
                                                unsigned short* __restrict__ Qo,
                                                unsigned short* __restrict__ Ko,
                                                unsigned short* __restrict__ Vt) {
  __shared__ unsigned short As[64 * 32];        // [row][k] bf16
  __shared__ unsigned short Bs[3][64 * 32];     // [col][k] bf16 (transposed W)
  const int t    = threadIdx.x;
  const int wave = t >> 5;
  const int lane = t & 31;
  const int row  = lane & 15;
  const int hi   = lane >> 4;
  const int Mbase = blockIdx.y * 64;
  const int Nbase = blockIdx.x * 64;

  v8f acc[3][4];
#pragma unroll
  for (int p = 0; p < 3; ++p)
#pragma unroll
    for (int c = 0; c < 4; ++c) acc[p][c] = kZero8;

  for (int k0 = 0; k0 < EMB; k0 += 32) {
    __syncthreads();
    // stage A: 64x32 f32 -> bf16, 1024 dwords, packed conversion
#pragma unroll
    for (int u = 0; u < 8; ++u) {
      int d = t + u * 128;
      int r = d >> 4, kd = d & 15;
      const float* xp = &X[(size_t)(Mbase + r) * EMB + k0 + 2 * kd];
      v2f x2 = *(const v2f*)xp;
      ((unsigned int*)As)[r * 16 + kd] = pack_bf16(x2.x, x2.y);
    }
    // stage B tiles (transposed) for all three projections
#pragma unroll
    for (int p = 0; p < 3; ++p) {
      const float* Wp = (p == 0) ? Wq : (p == 1) ? Wk : Wv;
#pragma unroll
      for (int u = 0; u < 8; ++u) {
        int d = t + u * 128;
        int col = d & 63, kp = d >> 6;
        const float* wp = &Wp[(size_t)(k0 + 2 * kp) * EMB + Nbase + col];
        ((unsigned int*)Bs[p])[col * 16 + kp] = pack_bf16(wp[0], wp[EMB]);
      }
    }
    if (k0 + 32 < EMB) {   // global_prefetch_b8 of next K-slab
      __builtin_prefetch(&X[(size_t)(Mbase + (t >> 3)) * EMB + k0 + 32], 0, 1);
      __builtin_prefetch(&Wq[(size_t)(k0 + 32 + (t >> 2)) * EMB + Nbase], 0, 1);
    }
    __syncthreads();

    Frag a;
    const v4u* ar = (const v4u*)(As + (wave * 16 + row) * 32);
    a.q[0] = ar[hi];
    a.q[1] = ar[hi + 2];
#pragma unroll
    for (int p = 0; p < 3; ++p) {
#pragma unroll
      for (int c = 0; c < 4; ++c) {
        Frag bb;
        const v4u* br = (const v4u*)(Bs[p] + (c * 16 + row) * 32);
        bb.q[0] = br[hi];
        bb.q[1] = br[hi + 2];
        acc[p][c] = wmma_bf16(a, bb, acc[p][c]);
      }
    }
  }

  // epilogue: bias + bf16 stores
#pragma unroll
  for (int p = 0; p < 3; ++p) {
    const float* bp = (p == 0) ? bq : (p == 1) ? bk : bv;
#pragma unroll
    for (int c = 0; c < 4; ++c) {
      float bvv = bp[Nbase + c * 16 + row];
#pragma unroll
      for (int r = 0; r < 8; ++r) {
        int m = Mbase + wave * 16 + r + 8 * hi;
        int e = Nbase + c * 16 + row;
        int b = m >> 11, n = m & (SEQ - 1);
        int h = e >> 6,  d = e & (HD - 1);
        unsigned short val = f2bf(acc[p][c][r] + bvv);
        if (p == 0)      Qo[(((size_t)(b * HEADS + h)) * SEQ + n) * HD + d] = val;
        else if (p == 1) Ko[(((size_t)(b * HEADS + h)) * SEQ + n) * HD + d] = val;
        else             Vt[(((size_t)(b * HEADS + h)) * HD + d) * SEQ + n] = val;
      }
    }
  }
}

// ---------------------------------------------------------------------------
// Kernel 2: sliding-window flash attention. One wave per 16-query tile.
// Q,K bf16 [B,H,N,D]; V bf16 transposed [B,H,D,N]; out bf16 [B,N,E].
// ---------------------------------------------------------------------------
__global__ __launch_bounds__(256) void attn(const unsigned short* __restrict__ Q,
                                            const unsigned short* __restrict__ K,
                                            const unsigned short* __restrict__ Vt,
                                            unsigned short* __restrict__ AO) {
  __shared__ unsigned short Pb[8][16 * 32];   // per-wave P repack tile
  const int wave = threadIdx.x >> 5;
  const int lane = threadIdx.x & 31;
  const int row  = lane & 15;
  const int hi   = lane >> 4;
  const int gw   = blockIdx.x * 8 + wave;     // 4096 waves total
  const int bh   = gw >> 7;                   // [0, 32)
  const int i0   = (gw & 127) * 16;

  Frag aq0, aq1;
  {
    const v4u* qp = (const v4u*)(Q + ((size_t)bh * SEQ + i0 + row) * HD);
    aq0.q[0] = qp[hi + 0]; aq0.q[1] = qp[hi + 2];   // D 0..31
    aq1.q[0] = qp[hi + 4]; aq1.q[1] = qp[hi + 6];   // D 32..63
  }

  float m_i[8], l_i[8];
  v8f Oc[4] = {kZero8, kZero8, kZero8, kZero8};
#pragma unroll
  for (int r = 0; r < 8; ++r) { m_i[r] = -3.0e38f; l_i[r] = 0.f; }

  const float scale = 0.125f;   // 1/sqrt(64)
  int jstart = i0 - WIN; if (jstart < 0) jstart = 0;
  int jend   = i0 + 16 + WIN; if (jend > SEQ) jend = SEQ;

  for (int jc = jstart; jc < jend; jc += 32) {
    // ---- S = Q @ K^T for two 16-key half-tiles ----
    Frag bk[2][2];
#pragma unroll
    for (int h2 = 0; h2 < 2; ++h2) {
      int j = jc + h2 * 16 + row;
      int jcl = j >= SEQ ? SEQ - 1 : j;
      const v4u* kp = (const v4u*)(K + ((size_t)bh * SEQ + jcl) * HD);
      bk[h2][0].q[0] = kp[hi + 0]; bk[h2][0].q[1] = kp[hi + 2];
      bk[h2][1].q[0] = kp[hi + 4]; bk[h2][1].q[1] = kp[hi + 6];
    }
    v8f S0 = kZero8, S1 = kZero8;
    S0 = wmma_bf16(aq0, bk[0][0], S0);
    S0 = wmma_bf16(aq1, bk[0][1], S0);
    S1 = wmma_bf16(aq0, bk[1][0], S1);
    S1 = wmma_bf16(aq1, bk[1][1], S1);

    // ---- masked online softmax (row stats across 16-lane groups) ----
    float p0[8], p1[8];
#pragma unroll
    for (int r = 0; r < 8; ++r) {
      int ig  = i0 + r + 8 * hi;
      int jj0 = jc + row;
      int jj1 = jc + 16 + row;
      float s0 = S0[r] * scale;
      float s1 = S1[r] * scale;
      if (!((jj0 < SEQ) && (ig - jj0 <= WIN) && (jj0 - ig <= WIN))) s0 = -3.0e38f;
      if (!((jj1 < SEQ) && (ig - jj1 <= WIN) && (jj1 - ig <= WIN))) s1 = -3.0e38f;
      float mx = fmaxf(s0, s1);
#pragma unroll
      for (int d = 1; d < 16; d <<= 1) mx = fmaxf(mx, __shfl_xor(mx, d, 32));
      float mn = fmaxf(m_i[r], mx);
      float al = __expf(m_i[r] - mn);
      p0[r] = __expf(s0 - mn);
      p1[r] = __expf(s1 - mn);
      float rs = p0[r] + p1[r];
#pragma unroll
      for (int d = 1; d < 16; d <<= 1) rs += __shfl_xor(rs, d, 32);
      l_i[r] = l_i[r] * al + rs;
      m_i[r] = mn;
      Oc[0][r] *= al; Oc[1][r] *= al; Oc[2][r] *= al; Oc[3][r] *= al;
    }

    // ---- repack P (C layout) -> A layout via LDS ----
    unsigned short* pb = Pb[wave];
#pragma unroll
    for (int r = 0; r < 8; ++r) {
      int M = r + 8 * hi;
      pb[M * 32 + row]      = f2bf(p0[r]);
      pb[M * 32 + 16 + row] = f2bf(p1[r]);
    }
    asm volatile("s_wait_dscnt 0x0" ::: "memory");
    Frag ap;
    {
      const v4u* pp = (const v4u*)(pb + row * 32);
      ap.q[0] = pp[hi];
      ap.q[1] = pp[hi + 2];
    }
    asm volatile("s_wait_dscnt 0x0" ::: "memory");

    // ---- O += P @ V : B-frag dwords = adjacent key pairs from V^T ----
#pragma unroll
    for (int c = 0; c < 4; ++c) {
      Frag bv;
      int dcol = c * 16 + row;
      const unsigned int* vrow =
          (const unsigned int*)(Vt + ((size_t)bh * HD + dcol) * SEQ);
#pragma unroll
      for (int d8 = 0; d8 < 8; ++d8) {
        int k2 = ((d8 & 4) ? d8 + 4 : d8) + hi * 4;   // key-pair index 0..15
        int j0 = jc + 2 * k2;
        if (j0 > SEQ - 2) j0 = SEQ - 2;
        bv.u[d8] = vrow[j0 >> 1];
      }
      Oc[c] = wmma_bf16(ap, bv, Oc[c]);
    }
  }

  // ---- normalize + store to [B,N,E] bf16 ----
  const int b = bh >> 4, h = bh & 15;
#pragma unroll
  for (int c = 0; c < 4; ++c) {
#pragma unroll
    for (int r = 0; r < 8; ++r) {
      int n = i0 + r + 8 * hi;
      int e = h * HD + c * 16 + row;
      AO[((size_t)b * SEQ + n) * EMB + e] = f2bf(Oc[c][r] / l_i[r]);
    }
  }
}

// ---------------------------------------------------------------------------
// Kernel 3: out = AO(bf16) @ Wo + bo (f32 out). A tile staged with async
// global->LDS copies when the toolchain exposes the builtin.
// ---------------------------------------------------------------------------
__global__ __launch_bounds__(128) void out_gemm(const unsigned short* __restrict__ Ain,
                                                const float* __restrict__ W,
                                                const float* __restrict__ bias,
                                                float* __restrict__ out) {
  __shared__ unsigned short As[64 * 32];
  __shared__ unsigned short Bs[64 * 32];
  const int t    = threadIdx.x;
  const int wave = t >> 5;
  const int lane = t & 31;
  const int row  = lane & 15;
  const int hi   = lane >> 4;
  const int Mbase = blockIdx.y * 64;
  const int Nbase = blockIdx.x * 64;

  v8f acc[4] = {kZero8, kZero8, kZero8, kZero8};

  for (int k0 = 0; k0 < EMB; k0 += 32) {
    __syncthreads();
#if defined(__has_builtin) && __has_builtin(__builtin_amdgcn_global_load_async_to_lds_b128)
    // 64 rows x 64B -> 256 x b128 async transfers, 2 per thread
#pragma unroll
    for (int u = 0; u < 2; ++u) {
      int d = t + u * 128;
      int r = d >> 2, seg = d & 3;
      __builtin_amdgcn_global_load_async_to_lds_b128(
          (gv4i*)(Ain + (size_t)(Mbase + r) * EMB + k0 + seg * 8),
          (lv4i*)(As + r * 32 + seg * 8),
          0, 0);
    }
#else
#pragma unroll
    for (int u = 0; u < 8; ++u) {
      int d = t + u * 128;
      int r = d >> 4, kd = d & 15;
      ((unsigned int*)As)[r * 16 + kd] =
          ((const unsigned int*)(Ain + (size_t)(Mbase + r) * EMB + k0))[kd];
    }
#endif
#pragma unroll
    for (int u = 0; u < 8; ++u) {
      int d = t + u * 128;
      int col = d & 63, kp = d >> 6;
      const float* wp = &W[(size_t)(k0 + 2 * kp) * EMB + Nbase + col];
      ((unsigned int*)Bs)[col * 16 + kp] = pack_bf16(wp[0], wp[EMB]);
    }
    if (k0 + 32 < EMB)
      __builtin_prefetch(&W[(size_t)(k0 + 32 + (t >> 2)) * EMB + Nbase], 0, 1);
#if defined(__has_builtin) && __has_builtin(__builtin_amdgcn_global_load_async_to_lds_b128)
    asm volatile("s_wait_asynccnt 0x0" ::: "memory");
#endif
    __syncthreads();

    Frag a;
    const v4u* ar = (const v4u*)(As + (wave * 16 + row) * 32);
    a.q[0] = ar[hi];
    a.q[1] = ar[hi + 2];
#pragma unroll
    for (int c = 0; c < 4; ++c) {
      Frag bb;
      const v4u* br = (const v4u*)(Bs + (c * 16 + row) * 32);
      bb.q[0] = br[hi];
      bb.q[1] = br[hi + 2];
      acc[c] = wmma_bf16(a, bb, acc[c]);
    }
  }

#pragma unroll
  for (int c = 0; c < 4; ++c) {
    float bvv = bias[Nbase + c * 16 + row];
#pragma unroll
    for (int r = 0; r < 8; ++r) {
      int m = Mbase + wave * 16 + r + 8 * hi;
      int e = Nbase + c * 16 + row;
      out[(size_t)m * EMB + e] = acc[c][r] + bvv;
    }
  }
}

__global__ void iota_kernel(int* __restrict__ p) {
  int i = blockIdx.x * 256 + threadIdx.x;
  if (i < SEQ) p[i] = i;
}

// ---------------------------------------------------------------------------
extern "C" void kernel_launch(void* const* d_in, const int* in_sizes, int n_in,
                              void* d_out, int out_size, void* d_ws, size_t ws_size,
                              hipStream_t stream) {
  (void)in_sizes; (void)n_in; (void)out_size; (void)ws_size;
  const float* q    = (const float*)d_in[0];
  const float* Wq_w = (const float*)d_in[1];
  const float* Wq_b = (const float*)d_in[2];
  const float* Wk_w = (const float*)d_in[3];
  const float* Wk_b = (const float*)d_in[4];
  const float* Wv_w = (const float*)d_in[5];
  const float* Wv_b = (const float*)d_in[6];
  const float* Wo_w = (const float*)d_in[7];
  const float* Wo_b = (const float*)d_in[8];

  const size_t PER = (size_t)BATCH * HEADS * SEQ * HD;   // 4,194,304 bf16 elems
  unsigned short* Qb = (unsigned short*)d_ws;
  unsigned short* Kb = Qb + PER;
  unsigned short* Vt = Kb + PER;
  unsigned short* Ab = Vt + PER;

  dim3 gg(EMB / 64, (BATCH * SEQ) / 64);   // (16, 64)
  qkv_gemm<<<gg, 128, 0, stream>>>(q, Wq_w, Wk_w, Wv_w, Wq_b, Wk_b, Wv_b,
                                   Qb, Kb, Vt);

  attn<<<(BATCH * HEADS * (SEQ / 16)) / 8, 256, 0, stream>>>(Qb, Kb, Vt, Ab);

  out_gemm<<<gg, 128, 0, stream>>>(Ab, Wo_w, Wo_b, (float*)d_out);

  iota_kernel<<<(SEQ + 255) / 256, 256, 0, stream>>>(
      ((int*)d_out) + (size_t)BATCH * SEQ * EMB);
}
